// ArnDiffRnnAgent_59785944761088
// MI455X (gfx1250) — compile-verified, matrix operand order checked by reading, not run on hardware
//
#include <hip/hip_runtime.h>
#include <hip/hip_bf16.h>
#include <stdint.h>

typedef __bf16 bf16;
typedef __attribute__((ext_vector_type(16))) __bf16 v16bf;
typedef __attribute__((ext_vector_type(8)))  __bf16 v8bf;
typedef __attribute__((ext_vector_type(8)))  float  v8f;

#define B_SZ 8192

// ---------------------------------------------------------------------------
// WMMA helpers (CDNA5 wave32, v_wmma_f32_16x16x32_bf16)
// ---------------------------------------------------------------------------
__device__ __forceinline__ v8f wmma_bf16(v16bf a, v16bf b, v8f c) {
  // 8 args: (neg_a, A, neg_b, B, c_mod, C, reuse_a, reuse_b)
  return __builtin_amdgcn_wmma_f32_16x16x32_bf16(false, a, false, b, (short)0, c,
                                                 false, false);
}

// A/B fragment load for 16x16x32 bf16 (ISA 7.12.2 16-bit A layout):
// lane L (row = L%16): lanes 0-15 hold K = {k0..k0+7, k0+16..k0+23},
// lanes 16-31 hold the +8 shifted set. Two 16B loads per lane.
__device__ __forceinline__ v16bf load_frag(const bf16* __restrict__ row, int k0,
                                           int lane) {
  int kb = k0 + ((lane & 16) ? 8 : 0);
  v8bf lo = *(const v8bf*)(row + kb);
  v8bf hi = *(const v8bf*)(row + kb + 16);
  v16bf r;
#pragma unroll
  for (int i = 0; i < 8; ++i) { r[i] = lo[i]; r[i + 8] = hi[i]; }
  return r;
}

// generic strided-view row mapping: element offset = (i/g)*s1 + (i%g)*s2 + off
__device__ __forceinline__ long map_row(long i, int g, long s1, long s2, long off) {
  long q = i / g;
  return q * s1 + (i - q * g) * s2 + off;
}

// ---------------------------------------------------------------------------
// f32 -> bf16 conversion (grid-stride)
// ---------------------------------------------------------------------------
__global__ void cvt_f32_bf16_kernel(const float* __restrict__ src,
                                    bf16* __restrict__ dst, long n) {
  long i = (long)blockIdx.x * blockDim.x + threadIdx.x;
  long stride = (long)gridDim.x * blockDim.x;
  for (; i < n; i += stride) dst[i] = (bf16)src[i];
}

// ---------------------------------------------------------------------------
// GEMM: C[M x 256] = act(A[M x K] @ W[256 x K]^T + bias)
// A rows come through the generic mapping; W is row-major (N x K) bf16.
// Block = 256 threads = 8 waves arranged 4(M) x 2(N); block tile 128 x 64.
// Each wave: 32x32 tile = 4 WMMA accumulators.
// ---------------------------------------------------------------------------
__global__ __launch_bounds__(256) void gemm_bf16_kernel(
    const bf16* __restrict__ A, int ag, long as1, long as2, long aoff,
    const bf16* __restrict__ W, const float* __restrict__ bias,
    float* __restrict__ Cf, bf16* __restrict__ Cb, int K, int relu) {
  const int lane = threadIdx.x & 31;
  const int wave = threadIdx.x >> 5;
  const int l15 = lane & 15;
  const long mBase = (long)blockIdx.x * 128 + (long)(wave >> 1) * 32;
  const int nBase = blockIdx.y * 64 + (wave & 1) * 32;

  const bf16* a0 = A + map_row(mBase + l15, ag, as1, as2, aoff);
  const bf16* a1 = A + map_row(mBase + 16 + l15, ag, as1, as2, aoff);
  const bf16* w0 = W + (long)(nBase + l15) * K;
  const bf16* w1 = W + (long)(nBase + 16 + l15) * K;

  v8f c00 = {}, c01 = {}, c10 = {}, c11 = {};
  for (int k0 = 0; k0 < K; k0 += 32) {
    v16bf fa0 = load_frag(a0, k0, lane);
    v16bf fa1 = load_frag(a1, k0, lane);
    v16bf fb0 = load_frag(w0, k0, lane);
    v16bf fb1 = load_frag(w1, k0, lane);
    c00 = wmma_bf16(fa0, fb0, c00);
    c01 = wmma_bf16(fa0, fb1, c01);
    c10 = wmma_bf16(fa1, fb0, c10);
    c11 = wmma_bf16(fa1, fb1, c11);
  }

  const int mOff = (lane >> 4) * 8;  // C/D layout: lanes 16-31 hold M+8
  const float b0 = bias[nBase + l15];
  const float b1 = bias[nBase + 16 + l15];
#pragma unroll
  for (int v = 0; v < 8; ++v) {
    long r0 = mBase + mOff + v;
    long r1 = r0 + 16;
    float v00 = c00[v] + b0, v01 = c01[v] + b1;
    float v10 = c10[v] + b0, v11 = c11[v] + b1;
    if (relu) {
      v00 = fmaxf(v00, 0.f); v01 = fmaxf(v01, 0.f);
      v10 = fmaxf(v10, 0.f); v11 = fmaxf(v11, 0.f);
    }
    long i00 = r0 * 256 + nBase + l15;
    long i01 = r0 * 256 + nBase + 16 + l15;
    long i10 = r1 * 256 + nBase + l15;
    long i11 = r1 * 256 + nBase + 16 + l15;
    if (Cf) { Cf[i00] = v00; Cf[i01] = v01; Cf[i10] = v10; Cf[i11] = v11; }
    if (Cb) {
      Cb[i00] = (bf16)v00; Cb[i01] = (bf16)v01;
      Cb[i10] = (bf16)v10; Cb[i11] = (bf16)v11;
    }
  }
}

// ---------------------------------------------------------------------------
// Fused GRU cell: h' = (1-z)*tanh(gin + r*ghn) + z*h_prev with
//   gi = x @ Wih^T + bih,  gh = h @ Whh^T + bhh  (Wih/Whh are 768 x 256)
// Never materializes gi/gh: each wave holds 6 accumulators (r/z/n for gi,gh)
// for one 16x16 output tile, fusing both GEMMs + gates + scatter-store.
// Block = 256 threads = 8 waves = 8 column tiles; grid = (M/16, 2).
// ---------------------------------------------------------------------------
__global__ __launch_bounds__(256) void gru_fused_kernel(
    const bf16* __restrict__ Xb,                               // M x 256, contiguous
    const bf16* __restrict__ Hb, int hg, long hs1, long hs2, long hoff,
    const bf16* __restrict__ Wih, const bf16* __restrict__ Whh,
    const float* __restrict__ bih, const float* __restrict__ bhh,
    const float* __restrict__ Hprev,                           // f32, same mapping
    float* __restrict__ Out) {                                 // f32, same mapping
  const int lane = threadIdx.x & 31;
  const int wave = threadIdx.x >> 5;
  const int l15 = lane & 15;
  const long mBase = (long)blockIdx.x * 16;
  const int c = (blockIdx.y * 8 + wave) * 16;  // column tile base, 0..240

  const long iRow = mBase + l15;
  const bf16* xrow = Xb + iRow * 256;
  const bf16* hrow = Hb + map_row(iRow, hg, hs1, hs2, hoff);
  const bf16* wir = Wih + (long)(c + l15) * 256;
  const bf16* wiz = Wih + (long)(256 + c + l15) * 256;
  const bf16* win = Wih + (long)(512 + c + l15) * 256;
  const bf16* whr = Whh + (long)(c + l15) * 256;
  const bf16* whz = Whh + (long)(256 + c + l15) * 256;
  const bf16* whn = Whh + (long)(512 + c + l15) * 256;

  v8f aIr = {}, aIz = {}, aIn = {}, aHr = {}, aHz = {}, aHn = {};
  for (int k0 = 0; k0 < 256; k0 += 32) {
    v16bf ax = load_frag(xrow, k0, lane);
    v16bf ah = load_frag(hrow, k0, lane);
    aIr = wmma_bf16(ax, load_frag(wir, k0, lane), aIr);
    aIz = wmma_bf16(ax, load_frag(wiz, k0, lane), aIz);
    aIn = wmma_bf16(ax, load_frag(win, k0, lane), aIn);
    aHr = wmma_bf16(ah, load_frag(whr, k0, lane), aHr);
    aHz = wmma_bf16(ah, load_frag(whz, k0, lane), aHz);
    aHn = wmma_bf16(ah, load_frag(whn, k0, lane), aHn);
  }

  const int col = c + l15;
  const float bir = bih[col],       bhr = bhh[col];
  const float biz = bih[256 + col], bhz = bhh[256 + col];
  const float bin = bih[512 + col], bhn = bhh[512 + col];
  const int mOff = (lane >> 4) * 8;
#pragma unroll
  for (int v = 0; v < 8; ++v) {
    long i = mBase + mOff + v;
    long ho = map_row(i, hg, hs1, hs2, hoff);
    float r = 1.0f / (1.0f + __expf(-((aIr[v] + bir) + (aHr[v] + bhr))));
    float z = 1.0f / (1.0f + __expf(-((aIz[v] + biz) + (aHz[v] + bhz))));
    float n = tanhf((aIn[v] + bin) + r * (aHn[v] + bhn));
    float hp = Hprev[ho + col];
    Out[ho + col] = (1.0f - z) * n + z * hp;
  }
}

// ---------------------------------------------------------------------------
// wo_q: q[b, 0:6] = h_env[b] @ W_wo^T + b_wo   (tiny: plain f32 dots)
// ---------------------------------------------------------------------------
__global__ void woq_kernel(const float* __restrict__ hiddenOut,
                           const float* __restrict__ Wwo,
                           const float* __restrict__ bwo, float* __restrict__ q) {
  int idx = blockIdx.x * blockDim.x + threadIdx.x;
  if (idx >= B_SZ * 6) return;
  int b = idx / 6, j = idx % 6;
  const float* h = hiddenOut + (long)b * 4352;
  const float* w = Wwo + j * 256;
  float s = bwo[j];
#pragma unroll 4
  for (int k = 0; k < 256; ++k) s += h[k] * w[k];
  q[(long)b * 22 + j] = s;
}

// ---------------------------------------------------------------------------
// attack_q: q[b, 6+n] = dot(env_h2[b], h3[b, n])
// ---------------------------------------------------------------------------
__global__ void attack_kernel(const float* __restrict__ envh2,
                              const float* __restrict__ hiddenOut,
                              float* __restrict__ q) {
  int idx = blockIdx.x * blockDim.x + threadIdx.x;
  if (idx >= B_SZ * 16) return;
  int b = idx >> 4, n = idx & 15;
  const float* x = envh2 + (long)b * 256;
  const float* h = hiddenOut + (long)b * 4352 + (long)(1 + n) * 256;
  float s = 0.f;
#pragma unroll 4
  for (int k = 0; k < 256; ++k) s += x[k] * h[k];
  q[(long)b * 22 + 6 + n] = s;
}

// ---------------------------------------------------------------------------
extern "C" void kernel_launch(void* const* d_in, const int* in_sizes, int n_in,
                              void* d_out, int out_size, void* d_ws, size_t ws_size,
                              hipStream_t stream) {
  const float* inputs  = (const float*)d_in[0];
  const float* hidden  = (const float*)d_in[1];
  const float* W_env1  = (const float*)d_in[2];
  const float* b_env1  = (const float*)d_in[3];
  const float* W_env2  = (const float*)d_in[4];
  const float* b_env2  = (const float*)d_in[5];
  const float* wih_env = (const float*)d_in[6];
  const float* whh_env = (const float*)d_in[7];
  const float* bih_env = (const float*)d_in[8];
  const float* bhh_env = (const float*)d_in[9];
  const float* W_wo    = (const float*)d_in[10];
  const float* b_wo    = (const float*)d_in[11];
  const float* W_e1a   = (const float*)d_in[12];
  const float* b_e1a   = (const float*)d_in[13];
  const float* W_e2a   = (const float*)d_in[14];
  const float* b_e2a   = (const float*)d_in[15];
  const float* wih_ea  = (const float*)d_in[16];
  const float* whh_ea  = (const float*)d_in[17];
  const float* bih_ea  = (const float*)d_in[18];
  const float* bhh_ea  = (const float*)d_in[19];
  const float* W_e1b   = (const float*)d_in[20];
  const float* b_e1b   = (const float*)d_in[21];
  const float* W_e2b   = (const float*)d_in[22];
  const float* b_e2b   = (const float*)d_in[23];
  const float* wih_eb  = (const float*)d_in[24];
  const float* whh_eb  = (const float*)d_in[25];
  const float* bih_eb  = (const float*)d_in[26];
  const float* bhh_eb  = (const float*)d_in[27];

  float* q    = (float*)d_out;
  float* hOut = q + (long)B_SZ * 22;  // (B, 17*256)

  char* ws = (char*)d_ws;
  size_t off = 0;
  auto alloc = [&](size_t bytes) -> char* {
    off = (off + 255) & ~(size_t)255;
    char* p = ws + off;
    off += bytes;
    return p;
  };

  bf16* Xb    = (bf16*)alloc((size_t)B_SZ * 1504 * 2);
  bf16* Hb    = (bf16*)alloc((size_t)B_SZ * 4352 * 2);
  bf16* eh1b  = (bf16*)alloc((size_t)B_SZ * 256 * 2);
  float* eh2f = (float*)alloc((size_t)B_SZ * 256 * 4);
  bf16* eh2b  = (bf16*)alloc((size_t)B_SZ * 256 * 2);
  bf16* eA    = (bf16*)alloc((size_t)65536 * 256 * 2);  // e1 (reused a/b)
  bf16* eBuf  = (bf16*)alloc((size_t)65536 * 256 * 2);  // e2 (reused a/b)
  bf16* Wenv1b  = (bf16*)alloc((size_t)256 * 1504 * 2);
  bf16* Wenv2b  = (bf16*)alloc((size_t)256 * 256 * 2);
  bf16* wihEnvB = (bf16*)alloc((size_t)768 * 256 * 2);
  bf16* whhEnvB = (bf16*)alloc((size_t)768 * 256 * 2);
  bf16* We1aB   = (bf16*)alloc((size_t)256 * 64 * 2);
  bf16* We2aB   = (bf16*)alloc((size_t)256 * 256 * 2);
  bf16* wihEaB  = (bf16*)alloc((size_t)768 * 256 * 2);
  bf16* whhEaB  = (bf16*)alloc((size_t)768 * 256 * 2);
  bf16* We1bB   = (bf16*)alloc((size_t)256 * 64 * 2);
  bf16* We2bB   = (bf16*)alloc((size_t)256 * 256 * 2);
  bf16* wihEbB  = (bf16*)alloc((size_t)768 * 256 * 2);
  bf16* whhEbB  = (bf16*)alloc((size_t)768 * 256 * 2);

  auto cvt = [&](const float* s, bf16* d, long n) {
    long blocks = (n + 255) / 256;
    if (blocks > 4096) blocks = 4096;
    cvt_f32_bf16_kernel<<<dim3((unsigned)blocks), dim3(256), 0, stream>>>(s, d, n);
  };
  auto gemm = [&](const bf16* A, int ag, long as1, long as2, long aoff,
                  const bf16* W, const float* bias, float* Cf, bf16* Cb, int M,
                  int K, int relu) {
    dim3 grid(M / 128, 4);  // N = 256 always: 4 column blocks of 64
    gemm_bf16_kernel<<<grid, dim3(256), 0, stream>>>(A, ag, as1, as2, aoff, W,
                                                     bias, Cf, Cb, K, relu);
  };
  auto gru = [&](const bf16* X, int hg, long hs1, long hs2, long hoff,
                 const bf16* Wih, const bf16* Whh, const float* bih,
                 const float* bhh, int M) {
    dim3 grid(M / 16, 2);  // 8 waves/block x 2 = 16 column tiles of 16
    gru_fused_kernel<<<grid, dim3(256), 0, stream>>>(X, Hb, hg, hs1, hs2, hoff,
                                                     Wih, Whh, bih, bhh, hidden,
                                                     hOut);
  };

  // ---- stage inputs/weights as bf16 ----
  cvt(inputs, Xb, (long)B_SZ * 1504);
  cvt(hidden, Hb, (long)B_SZ * 4352);
  cvt(W_env1, Wenv1b, 256L * 1504);
  cvt(W_env2, Wenv2b, 256L * 256);
  cvt(wih_env, wihEnvB, 768L * 256);
  cvt(whh_env, whhEnvB, 768L * 256);
  cvt(W_e1a, We1aB, 256L * 64);
  cvt(W_e2a, We2aB, 256L * 256);
  cvt(wih_ea, wihEaB, 768L * 256);
  cvt(whh_ea, whhEaB, 768L * 256);
  cvt(W_e1b, We1bB, 256L * 64);
  cvt(W_e2b, We2bB, 256L * 256);
  cvt(wih_eb, wihEbB, 768L * 256);
  cvt(whh_eb, whhEbB, 768L * 256);

  // ---- env trunk ----
  // env_h1 = relu(X @ W_env1^T + b); rows identity over X (lda 1504)
  gemm(Xb, 1, 1504, 0, 0, Wenv1b, b_env1, nullptr, eh1b, B_SZ, 1504, 1);
  // env_h2 = env_h1 @ W_env2^T + b (keep f32 for attack_q + bf16 for GRU)
  gemm(eh1b, 1, 256, 0, 0, Wenv2b, b_env2, eh2f, eh2b, B_SZ, 256, 0);
  // h_env = GRU(env_h2, hidden[:,0]); hidden row map: i*4352 + 0
  gru(eh2b, 1, 4352, 0, 0, wihEnvB, whhEnvB, bih_env, bhh_env, B_SZ);
  // wo_q
  woq_kernel<<<dim3((B_SZ * 6 + 255) / 256), dim3(256), 0, stream>>>(hOut, W_wo,
                                                                     b_wo, q);

  // ---- enemy branch a (rows i = b*8+n): feats at b*1504 + n*64 + 16,
  //      hidden at b*4352 + n*256 + 256 ----
  gemm(Xb, 8, 1504, 64, 16, We1aB, b_e1a, nullptr, eA, 65536, 64, 1);
  gemm(eA, 1, 256, 0, 0, We2aB, b_e2a, nullptr, eBuf, 65536, 256, 0);
  gru(eBuf, 8, 4352, 256, 256, wihEaB, whhEaB, bih_ea, bhh_ea, 65536);

  // ---- enemy branch b: feats offset 16+8*64=528, hidden offset (1+8)*256 ----
  gemm(Xb, 8, 1504, 64, 528, We1bB, b_e1b, nullptr, eA, 65536, 64, 1);
  gemm(eA, 1, 256, 0, 0, We2bB, b_e2b, nullptr, eBuf, 65536, 256, 0);
  gru(eBuf, 8, 4352, 256, 2304, wihEbB, whhEbB, bih_eb, bhh_eb, 65536);

  // ---- attack_q ----
  attack_kernel<<<dim3((B_SZ * 16) / 256), dim3(256), 0, stream>>>(eh2f, hOut, q);
}
